// NaturalAttention_32418413150218
// MI455X (gfx1250) — compile-verified, matrix operand order checked by reading, not run on hardware
//
#include <hip/hip_runtime.h>
#include <hip/hip_bf16.h>

// ---------------------------------------------------------------------------
// Problem constants (B,S,D,H from the reference)
// ---------------------------------------------------------------------------
#define B_   2
#define S_   2048
#define D_   1024
#define H_   16
#define HD_  64
#define M_   (B_ * S_)          // 4096 total rows

typedef __attribute__((ext_vector_type(16))) _Float16 v16h;
typedef __attribute__((ext_vector_type(8)))  _Float16 v8h;
typedef __attribute__((ext_vector_type(8)))  float    v8f;
typedef __attribute__((ext_vector_type(4)))  int      v4i;

// ---------------------------------------------------------------------------
// Optional CDNA5 async global->LDS DMA (ASYNCcnt path); guarded so the file
// compiles on toolchains without the builtin (fallback: load + ds_store).
// Probe round 3: builtin exists and takes (v4i AS1*, v4i AS3*, imm, imm).
// ---------------------------------------------------------------------------
#if defined(__has_builtin)
#if __has_builtin(__builtin_amdgcn_global_load_async_to_lds_b128)
#define HAVE_ASYNC_LDS 1
#endif
#endif
#ifndef HAVE_ASYNC_LDS
#define HAVE_ASYNC_LDS 0
#endif

__device__ __forceinline__ void wait_async0() {
#if defined(__has_builtin)
#if __has_builtin(__builtin_amdgcn_s_wait_asynccnt)
    __builtin_amdgcn_s_wait_asynccnt(0);
    return;
#endif
#endif
    asm volatile("s_wait_asynccnt 0x0" ::: "memory");
}

__device__ __forceinline__ void copy16_to_lds(const _Float16* __restrict__ g,
                                              _Float16* __restrict__ l) {
#if HAVE_ASYNC_LDS
    typedef __attribute__((address_space(1))) v4i gv4i;   // global int4
    typedef __attribute__((address_space(3))) v4i lv4i;   // LDS int4
    __builtin_amdgcn_global_load_async_to_lds_b128(
        (gv4i*)g, (lv4i*)(unsigned)(uintptr_t)l, 0, 0);
#else
    v8h t = *(const v8h*)g;
    *(v8h*)l = t;
#endif
}

// D = A(16x32 f16) * B(32x16 f16) + C(16x16 f32)
__device__ __forceinline__ v8f wmma_f16(v16h a, v16h b, v8f c) {
    return __builtin_amdgcn_wmma_f32_16x16x32_f16(
        /*neg_a=*/false, a, /*neg_b=*/false, b,
        /*c_mod=*/(short)0, c, /*reuse_a=*/false, /*reuse_b=*/false);
}

// Universal fragment loader for row-major f16 storage (global memory).
// CDNA5 16-bit layout: the lane's 16 halves cover
// K = {kbase + 8*(lane/16) + 0..7, +16..23}: two contiguous 16-byte chunks.
__device__ __forceinline__ v16h load_frag16(const _Float16* __restrict__ p,
                                            int row, int ld, int kbase) {
    const int lane = threadIdx.x & 31;
    const int hi   = lane >> 4;            // 0 or 1
    const _Float16* q = p + (size_t)row * ld + kbase + hi * 8;
    v8h lo = *(const v8h*)(q);
    v8h hv = *(const v8h*)(q + 16);
    v16h r;
#pragma unroll
    for (int i = 0; i < 8; ++i) { r[i] = lo[i]; r[i + 8] = hv[i]; }
    return r;
}

// Same fragment math, but reading a staged tile from LDS (ds_load_b128 pairs).
__device__ __forceinline__ v16h lds_frag16(const _Float16* p, int row, int ld,
                                           int kbase) {
    const int lane = threadIdx.x & 31;
    const int hi   = lane >> 4;
    const _Float16* q = p + row * ld + kbase + hi * 8;
    v8h lo = *(const v8h*)(q);
    v8h hv = *(const v8h*)(q + 16);
    v16h r;
#pragma unroll
    for (int i = 0; i < 8; ++i) { r[i] = lo[i]; r[i + 8] = hv[i]; }
    return r;
}

// ---------------------------------------------------------------------------
// Prep kernels
// ---------------------------------------------------------------------------
__global__ void cvt_f32_to_f16(const float* __restrict__ src,
                               _Float16* __restrict__ dst, int n) {
    int i = blockIdx.x * blockDim.x + threadIdx.x;
    if (i < n) dst[i] = (_Float16)src[i];
}

__global__ void transpose_w_f16(const float* __restrict__ W,
                                _Float16* __restrict__ WT) {
    int i = blockIdx.x * blockDim.x + threadIdx.x;   // i in [0, D*D)
    int k = i / D_;
    int n = i - k * D_;
    WT[(size_t)n * D_ + k] = (_Float16)W[i];         // coalesced read
}

// ---------------------------------------------------------------------------
// GEMM fragment set for a 32M x 64N wave tile at one 32-wide K step
// ---------------------------------------------------------------------------
struct GFrag {
    v16h a0, a1;             // rows mBase..+15, mBase+16..+31
    v16h b0, b1, b2, b3;     // cols nBase + {0,16,32,48}
};

__device__ __forceinline__ GFrag load_gemm_frags(const _Float16* __restrict__ X,
                                                 const _Float16* __restrict__ W,
                                                 int mBase, int nBase, int col,
                                                 int kb) {
    GFrag f;
    f.a0 = load_frag16(X, mBase + col,      D_, kb);
    f.a1 = load_frag16(X, mBase + 16 + col, D_, kb);
    f.b0 = load_frag16(W, nBase + col,      D_, kb);
    f.b1 = load_frag16(W, nBase + 16 + col, D_, kb);
    f.b2 = load_frag16(W, nBase + 32 + col, D_, kb);
    f.b3 = load_frag16(W, nBase + 48 + col, D_, kb);
    return f;
}

#define GEMM_8WMMA(F)                                   \
    do {                                                \
        acc[0] = wmma_f16((F).a0, (F).b0, acc[0]);      \
        acc[1] = wmma_f16((F).a0, (F).b1, acc[1]);      \
        acc[2] = wmma_f16((F).a0, (F).b2, acc[2]);      \
        acc[3] = wmma_f16((F).a0, (F).b3, acc[3]);      \
        acc[4] = wmma_f16((F).a1, (F).b0, acc[4]);      \
        acc[5] = wmma_f16((F).a1, (F).b1, acc[5]);      \
        acc[6] = wmma_f16((F).a1, (F).b2, acc[6]);      \
        acc[7] = wmma_f16((F).a1, (F).b3, acc[7]);      \
    } while (0)

// ---------------------------------------------------------------------------
// Fused QKV projection.  grid = (D/64, M/128, 3), block = 128 (4 waves).
// Each wave: 32x64 output tile, register double-buffered K-loop.
// ---------------------------------------------------------------------------
__global__ __launch_bounds__(128) void qkv_gemm(
    const _Float16* __restrict__ X, const _Float16* __restrict__ WT,
    const float* __restrict__ bq, const float* __restrict__ bk,
    const float* __restrict__ bv,
    _Float16* __restrict__ Qo, _Float16* __restrict__ Ko,
    _Float16* __restrict__ VTo) {
    const int lane = threadIdx.x & 31;
    const int wave = threadIdx.x >> 5;
    const int col  = lane & 15;
    const int hi   = lane >> 4;
    const int z     = blockIdx.z;
    const int mBase = (blockIdx.y * 4 + wave) * 32;
    const int nBase = blockIdx.x * 64;
    const _Float16* W = WT + (size_t)z * D_ * D_;
    const float* bias = (z == 0) ? bq : ((z == 1) ? bk : bv);

    v8f acc[8] = {};
    GFrag cur = load_gemm_frags(X, W, mBase, nBase, col, 0);
    for (int kb = 0; kb < D_; kb += 32) {
        const int kn = (kb + 32) & (D_ - 1);      // wraps to 0 on last iter
        GFrag nxt = load_gemm_frags(X, W, mBase, nBase, col, kn);
        // stream-prefetch the A panel ahead (global_prefetch_b8, near scope)
        __builtin_prefetch((const void*)(X + (size_t)(mBase + col) * D_ + kb + 256), 0, 3);
        GEMM_8WMMA(cur);
        cur = nxt;
    }

#pragma unroll
    for (int t = 0; t < 4; ++t) {
        const int n  = nBase + t * 16 + col;
        const int hh = n >> 6;           // / HD
        const int hd = n & (HD_ - 1);
        const float bsv = bias[n];
#pragma unroll
        for (int half = 0; half < 2; ++half) {
#pragma unroll
            for (int r = 0; r < 8; ++r) {
                const int m  = mBase + half * 16 + r + 8 * hi;  // D-layout row
                const int bb = m >> 11;                          // / S
                const int ss = m & (S_ - 1);
                const _Float16 hval = (_Float16)(acc[half * 4 + t][r] + bsv);
                if (z == 0)
                    Qo[((((size_t)bb * H_ + hh) * S_ + ss) * HD_) + hd] = hval;
                else if (z == 1)
                    Ko[((((size_t)bb * H_ + hh) * S_ + ss) * HD_) + hd] = hval;
                else
                    VTo[((((size_t)bb * H_ + hh) * HD_ + hd) * S_) + ss] = hval;
            }
        }
    }
}

// ---------------------------------------------------------------------------
// Flash attention with LDS-staged K/V.  grid = (S/64, H, B), block = 128
// (4 waves, 16 q-rows each).  All 4 waves consume the SAME 32-key K/V block,
// so one copy is staged per workgroup in LDS (async DMA when available),
// double-buffered: block i+1 streams in while block i is consumed.
//
// Scores computed TRANSPOSED: T = K_block(16x64) x Q^T(64x16); the C/D layout
// of T (lane = query col, vgpr = key row) matches the A-fragment layout of
// the following P·V WMMA exactly -> exp'd probs feed the next WMMA with no
// cross-lane movement.
// ---------------------------------------------------------------------------
#define KLD 72          // padded LDS row pitch for K tile (32 x 64 halves)
#define VLD 40          // padded LDS row pitch for V^T tile (64 x 32 halves)
#define NBLK (S_ / 32)

__global__ __launch_bounds__(128) void flash_attn(
    const _Float16* __restrict__ Q, const _Float16* __restrict__ K,
    const _Float16* __restrict__ VT, const float* __restrict__ mask,
    _Float16* __restrict__ CTX) {
    __shared__ _Float16 Kbuf[2][32 * KLD];
    __shared__ _Float16 Vbuf[2][HD_ * VLD];

    const int tid  = threadIdx.x;
    const int lane = tid & 31;
    const int wave = tid >> 5;
    const int col  = lane & 15;
    const int hi   = lane >> 4;
    const int b = blockIdx.z, h = blockIdx.y;
    const int qBase = (blockIdx.x * 4 + wave) * 16;

    const _Float16* Qh = Q  + ((size_t)b * H_ + h) * S_ * HD_;
    const _Float16* Kh = K  + ((size_t)b * H_ + h) * S_ * HD_;
    const _Float16* Vh = VT + ((size_t)b * H_ + h) * HD_ * S_;
    const float*    mk = mask + (size_t)b * S_;

    // Q as B-fragments (lane = query column), contraction over HD=64 -> 2 frags
    const v16h qb0 = load_frag16(Qh, qBase + col, HD_, 0);
    const v16h qb1 = load_frag16(Qh, qBase + col, HD_, 32);

    v8f o0 = {}, o1 = {}, o2 = {}, o3 = {};
    float mi = -3.0e38f, li = 0.0f;
    const float scale = 0.125f;            // 1/sqrt(64)

    // Cooperative stage of one 32-key block into LDS buffer `buf`.
    // 128 threads x 2 chunks x 16B cover each of the K and V tiles.
    auto stage = [&](int buf, int kb) {
#pragma unroll
        for (int i = 0; i < 2; ++i) {
            const int c  = tid + i * 128;            // 0..255
            const int kr = c >> 3, ko = (c & 7) * 8; // K: row 0..31, off 0..56
            const int vr = c >> 2, vo = (c & 3) * 8; // V: row 0..63, off 0..24
            copy16_to_lds(Kh + (size_t)(kb + kr) * HD_ + ko,
                          &Kbuf[buf][kr * KLD + ko]);
            copy16_to_lds(Vh + (size_t)vr * S_ + kb + vo,
                          &Vbuf[buf][vr * VLD + vo]);
        }
    };

    stage(0, 0);
    wait_async0();
    __syncthreads();

    for (int i = 0; i < NBLK; ++i) {
        const int kb  = i * 32;
        const int cb  = i & 1;
        if (i + 1 < NBLK) stage(cb ^ 1, kb + 32);   // next block in flight

        // T0: keys kb..kb+15, T1: keys kb+16..kb+31 (2 WMMAs each over HD)
        const _Float16* Kb = Kbuf[cb];
        v8f t0 = {}, t1 = {};
        t0 = wmma_f16(lds_frag16(Kb, col,      KLD, 0),  qb0, t0);
        t0 = wmma_f16(lds_frag16(Kb, col,      KLD, 32), qb1, t0);
        t1 = wmma_f16(lds_frag16(Kb, 16 + col, KLD, 0),  qb0, t1);
        t1 = wmma_f16(lds_frag16(Kb, 16 + col, KLD, 32), qb1, t1);

        // scale + additive mask (mask depends on key only), local row-max
        float mx = -3.0e38f;
#pragma unroll
        for (int r = 0; r < 8; ++r) {
            t0[r] = t0[r] * scale + mk[kb + r + 8 * hi];
            t1[r] = t1[r] * scale + mk[kb + 16 + r + 8 * hi];
            mx = fmaxf(mx, fmaxf(t0[r], t1[r]));
        }
        mx = fmaxf(mx, __shfl_xor(mx, 16, 32));    // combine the two key halves
        const float mnew  = fmaxf(mi, mx);
        const float alpha = __expf(mi - mnew);     // first iter: exp(-inf)=0
        float rs = 0.0f;
        v16h pf;                                   // P as A-fragment (16x32)
#pragma unroll
        for (int r = 0; r < 8; ++r) {
            const float p0 = __expf(t0[r] - mnew);
            const float p1 = __expf(t1[r] - mnew);
            rs += p0 + p1;
            pf[r]     = (_Float16)p0;              // keys r+8*hi
            pf[r + 8] = (_Float16)p1;              // keys 16+r+8*hi
        }
        rs += __shfl_xor(rs, 16, 32);
        li = li * alpha + rs;
        mi = mnew;
        // rescale O: O's D-layout has q = r+8*hi; alpha lives at lane==q
        float ar[8];
#pragma unroll
        for (int r = 0; r < 8; ++r) ar[r] = __shfl(alpha, r + 8 * hi, 32);
#pragma unroll
        for (int r = 0; r < 8; ++r) {
            o0[r] *= ar[r]; o1[r] *= ar[r]; o2[r] *= ar[r]; o3[r] *= ar[r];
        }
        // P·V from the staged V^T tile
        const _Float16* Vb = Vbuf[cb];
        o0 = wmma_f16(pf, lds_frag16(Vb,  0 + col, VLD, 0), o0);
        o1 = wmma_f16(pf, lds_frag16(Vb, 16 + col, VLD, 0), o1);
        o2 = wmma_f16(pf, lds_frag16(Vb, 32 + col, VLD, 0), o2);
        o3 = wmma_f16(pf, lds_frag16(Vb, 48 + col, VLD, 0), o3);

        wait_async0();        // own async stage complete before barrier
        __syncthreads();      // all waves done with buf[cb], next buf visible
    }

    const float linv = 1.0f / li;
    float lr[8];
#pragma unroll
    for (int r = 0; r < 8; ++r) lr[r] = __shfl(linv, r + 8 * hi, 32);
    // context back to (B, S, H*HD) row-major f16 for the output projection
#pragma unroll
    for (int r = 0; r < 8; ++r) {
        const int qrow = qBase + r + 8 * hi;
        const size_t rowoff = ((size_t)b * S_ + qrow) * D_ + (size_t)h * HD_;
        CTX[rowoff +  0 + col] = (_Float16)(o0[r] * lr[r]);
        CTX[rowoff + 16 + col] = (_Float16)(o1[r] * lr[r]);
        CTX[rowoff + 32 + col] = (_Float16)(o2[r] * lr[r]);
        CTX[rowoff + 48 + col] = (_Float16)(o3[r] * lr[r]);
    }
}

// ---------------------------------------------------------------------------
// Output projection: out = CTX @ Wo + bo, f32 result straight to d_out.
// ---------------------------------------------------------------------------
__global__ __launch_bounds__(128) void out_proj(
    const _Float16* __restrict__ CTX, const _Float16* __restrict__ WoT,
    const float* __restrict__ bo, float* __restrict__ out) {
    const int lane = threadIdx.x & 31;
    const int wave = threadIdx.x >> 5;
    const int col  = lane & 15;
    const int hi   = lane >> 4;
    const int mBase = (blockIdx.y * 4 + wave) * 32;
    const int nBase = blockIdx.x * 64;

    v8f acc[8] = {};
    GFrag cur = load_gemm_frags(CTX, WoT, mBase, nBase, col, 0);
    for (int kb = 0; kb < D_; kb += 32) {
        const int kn = (kb + 32) & (D_ - 1);
        GFrag nxt = load_gemm_frags(CTX, WoT, mBase, nBase, col, kn);
        __builtin_prefetch((const void*)(CTX + (size_t)(mBase + col) * D_ + kb + 256), 0, 3);
        GEMM_8WMMA(cur);
        cur = nxt;
    }

#pragma unroll
    for (int t = 0; t < 4; ++t) {
        const int n = nBase + t * 16 + col;
        const float bsv = bo[n];
#pragma unroll
        for (int half = 0; half < 2; ++half) {
#pragma unroll
            for (int r = 0; r < 8; ++r) {
                const int m = mBase + half * 16 + r + 8 * hi;
                out[(size_t)m * D_ + n] = acc[half * 4 + t][r] + bsv;
            }
        }
    }
}

// ---------------------------------------------------------------------------
// Launch: prep -> fused QKV GEMM -> flash attention -> output GEMM
// Workspace (f16 elems): [X16|CTX alias 4M][WT 4x1M][Q 4M][K 4M][VT 4M] ~40MB
// ---------------------------------------------------------------------------
extern "C" void kernel_launch(void* const* d_in, const int* in_sizes, int n_in,
                              void* d_out, int out_size, void* d_ws, size_t ws_size,
                              hipStream_t stream) {
    (void)in_sizes; (void)n_in; (void)out_size; (void)ws_size;
    const float* X    = (const float*)d_in[0];
    const float* mask = (const float*)d_in[1];
    const float* Wq = (const float*)d_in[2];
    const float* bq = (const float*)d_in[3];
    const float* Wk = (const float*)d_in[4];
    const float* bk = (const float*)d_in[5];
    const float* Wv = (const float*)d_in[6];
    const float* bv = (const float*)d_in[7];
    const float* Wo = (const float*)d_in[8];
    const float* bo = (const float*)d_in[9];
    float* out = (float*)d_out;

    const size_t MEG = 1024u * 1024u;
    _Float16* ws   = (_Float16*)d_ws;
    _Float16* X16  = ws;                 // 4M elems (reused as CTX after QKV)
    _Float16* WT   = ws + 4 * MEG;       // 4 x 1M elems (WqT,WkT,WvT,WoT)
    _Float16* Q16  = ws + 8 * MEG;       // 4M
    _Float16* K16  = ws + 12 * MEG;      // 4M
    _Float16* VT16 = ws + 16 * MEG;      // 4M
    _Float16* CTX  = X16;                // alias: X16 dead after qkv_gemm

    cvt_f32_to_f16<<<(M_ * D_) / 256, 256, 0, stream>>>(X, X16, M_ * D_);
    transpose_w_f16<<<(D_ * D_) / 256, 256, 0, stream>>>(Wq, WT + 0 * MEG);
    transpose_w_f16<<<(D_ * D_) / 256, 256, 0, stream>>>(Wk, WT + 1 * MEG);
    transpose_w_f16<<<(D_ * D_) / 256, 256, 0, stream>>>(Wv, WT + 2 * MEG);
    transpose_w_f16<<<(D_ * D_) / 256, 256, 0, stream>>>(Wo, WT + 3 * MEG);

    qkv_gemm<<<dim3(D_ / 64, M_ / 128, 3), 128, 0, stream>>>(
        X16, WT, bq, bk, bv, Q16, K16, VT16);

    flash_attn<<<dim3(S_ / 64, H_, B_), 128, 0, stream>>>(
        Q16, K16, VT16, mask, CTX);

    out_proj<<<dim3(D_ / 64, M_ / 128), 128, 0, stream>>>(
        CTX, WT + 3 * MEG, bo, out);
}
